// GlobalAdditiveAttention_1898375545100
// MI455X (gfx1250) — matrix operand order repaired and checked by reading, not compiled, not run on hardware
//
#include <hip/hip_runtime.h>
#include <cstdint>

#define HIDDEN 128
constexpr float TAU = 1.0f;

typedef __bf16 v16bf __attribute__((ext_vector_type(16)));
typedef float  v8f   __attribute__((ext_vector_type(8)));

__device__ __forceinline__ float fast_tanh(float x) {
    // tanh(x) = 1 - 2/(e^{2x}+1): one v_exp_f32 + one rcp, much cheaper than ocml tanh
    float e = __expf(2.0f * x);
    return 1.0f - 2.0f / (e + 1.0f);
}

__device__ __forceinline__ float waveSum(float v) {
#pragma unroll
    for (int d = 16; d >= 1; d >>= 1) v += __shfl_xor(v, d, 32);
    return v;
}
__device__ __forceinline__ float waveMax(float v) {
#pragma unroll
    for (int d = 16; d >= 1; d >>= 1) v = fmaxf(v, __shfl_xor(v, d, 32));
    return v;
}

// -----------------------------------------------------------------------------
// Kernel 1: s[i] = (tanh(h[i,:] @ W1^T + b1) . w2 + b2) / TAU  via bf16 WMMA.
// 8 waves/block, each wave computes a 16-row tile: D[16x128] = A[16x128] x B[128x128]
// as 4 K-steps of v_wmma_f32_16x16x32_bf16 across 8 N-tiles.
// -----------------------------------------------------------------------------
__global__ __launch_bounds__(256) void score_kernel(
    const float* __restrict__ h, const float* __restrict__ W1,
    const float* __restrict__ b1, const float* __restrict__ w2,
    const float* __restrict__ b2c, float* __restrict__ s, int n)
{
    __shared__ __align__(32) __bf16 W1s[HIDDEN * HIDDEN];   // 32 KB bf16 copy of W1

    const int tid = threadIdx.x;
    // cooperative fp32 -> bf16 convert of W1 into LDS (W1s[n*128+k] = W1[n][k])
    for (int i = tid; i < HIDDEN * HIDDEN; i += 256) W1s[i] = (__bf16)W1[i];
    __syncthreads();

    const int wave = tid >> 5;
    const int lane = tid & 31;
    const int lgrp = lane & 15;     // lane within 16-lane half
    const int hi   = lane >> 4;     // which half-wave (K-offset selector)
    const float b2v = b2c[0];

    // per-lane bias/weight for the 8 columns this lane owns in the C/D layout
    float b1v[8], w2v[8];
#pragma unroll
    for (int nt = 0; nt < 8; ++nt) {
        int col = nt * 16 + lgrp;
        b1v[nt] = b1[col];
        w2v[nt] = w2[col];
    }

    const int tilesTotal = n >> 4;                   // 16-row tiles (N % 16 == 0 path)
    for (int tb = blockIdx.x; tb * 8 < tilesTotal; tb += gridDim.x) {
        const int tile = tb * 8 + wave;
        if (tile < tilesTotal) {                     // wave-uniform: EXEC stays all-1s
            const int m0 = tile << 4;
            v8f acc[8] = {};                          // 8 N-tiles of 16x16 f32 accum

            // A rows: lane L<16 holds row m0+L, K = c..c+15 (c = hi*16)
            const float* hrow = h + (size_t)(m0 + lgrp) * HIDDEN + hi * 16;
#pragma unroll
            for (int kk = 0; kk < HIDDEN; kk += 32) {
                const float4* hp = (const float4*)(hrow + kk);
                float4 x0 = hp[0], x1 = hp[1];        // K = kk+c .. kk+c+7
                float4 y0 = hp[2], y1 = hp[3];        // K = kk+c+8 .. kk+c+15
                v16bf a;
                a[0]=(__bf16)x0.x; a[1]=(__bf16)x0.y; a[2]=(__bf16)x0.z; a[3]=(__bf16)x0.w;
                a[4]=(__bf16)x1.x; a[5]=(__bf16)x1.y; a[6]=(__bf16)x1.z; a[7]=(__bf16)x1.w;
                a[8]=(__bf16)y0.x; a[9]=(__bf16)y0.y; a[10]=(__bf16)y0.z; a[11]=(__bf16)y0.w;
                a[12]=(__bf16)y1.x; a[13]=(__bf16)y1.y; a[14]=(__bf16)y1.z; a[15]=(__bf16)y1.w;
#pragma unroll
                for (int nt = 0; nt < 8; ++nt) {
                    // B[k][n] = W1[n][k]; lane n<16 holds K=kk..kk+15 of col n,
                    // lanes 16-31 hold K=kk+16..kk+31 -> contiguous 32B LDS read
                    v16bf bm = *(const v16bf*)&W1s[(nt * 16 + lgrp) * HIDDEN + kk + hi * 16];
                    acc[nt] = __builtin_amdgcn_wmma_f32_16x16x32_bf16(
                        false, a, false, bm, (short)0, acc[nt], false, false);
                }
            }

            // epilogue: element (row m0 + r + 8*hi, col nt*16+lgrp) lives in acc[nt][r]
            float p[8] = {0.f, 0.f, 0.f, 0.f, 0.f, 0.f, 0.f, 0.f};
#pragma unroll
            for (int nt = 0; nt < 8; ++nt) {
#pragma unroll
                for (int r = 0; r < 8; ++r)
                    p[r] += fast_tanh(acc[nt][r] + b1v[nt]) * w2v[nt];
            }
#pragma unroll
            for (int r = 0; r < 8; ++r) {             // reduce across the 16-lane group
                float v = p[r];
                v += __shfl_xor(v, 1, 32);
                v += __shfl_xor(v, 2, 32);
                v += __shfl_xor(v, 4, 32);
                v += __shfl_xor(v, 8, 32);
                if (lgrp == 0) s[m0 + (hi << 3) + r] = (v + b2v) * (1.0f / TAU);
            }
        }
    }
}

// scalar fallback for a (<16)-row tail so any N is handled
__global__ __launch_bounds__(128) void score_tail_kernel(
    const float* __restrict__ h, const float* __restrict__ W1,
    const float* __restrict__ b1, const float* __restrict__ w2,
    const float* __restrict__ b2c, float* __restrict__ s, int row0)
{
    __shared__ float red[4];
    const int row = row0 + blockIdx.x;
    const int c = threadIdx.x;                  // one output column per thread
    const float* hr = h + (size_t)row * HIDDEN;
    float d = 0.f;
    for (int k = 0; k < HIDDEN; ++k) d = fmaf(hr[k], W1[c * HIDDEN + k], d);
    float t = fast_tanh(d + b1[c]) * w2[c];
    t = waveSum(t);
    if ((c & 31) == 0) red[c >> 5] = t;
    __syncthreads();
    if (c == 0) s[row] = (red[0] + red[1] + red[2] + red[3] + b2c[0]) * (1.0f / TAU);
}

// -----------------------------------------------------------------------------
// Kernel 2: one block per graph. Binary-search row range in sorted int64 batch,
// stable segment softmax over s, then weighted coalesced scatter-sum of h rows.
// -----------------------------------------------------------------------------
__global__ __launch_bounds__(HIDDEN) void pool_kernel(
    const float* __restrict__ h, const long long* __restrict__ batch,
    const float* __restrict__ s, float* __restrict__ out, int n)
{
    __shared__ float red[4];
    __shared__ float wbuf[HIDDEN];
    __shared__ int range[2];

    const int g   = blockIdx.x;
    const int tid = threadIdx.x;
    const int wave = tid >> 5, lane = tid & 31;

    if (tid < 2) {                               // lower_bound(g) and lower_bound(g+1)
        const long long target = (long long)g + tid;
        int lo = 0, hi = n;
        while (lo < hi) {
            int mid = (lo + hi) >> 1;
            if (batch[mid] < target) lo = mid + 1; else hi = mid;
        }
        range[tid] = lo;
    }
    __syncthreads();
    const int start = range[0], end = range[1];

    // --- segment max ---
    float m = -3.4e38f;
    for (int i = start + tid; i < end; i += HIDDEN) m = fmaxf(m, s[i]);
    m = waveMax(m);
    if (lane == 0) red[wave] = m;
    __syncthreads();
    m = fmaxf(fmaxf(red[0], red[1]), fmaxf(red[2], red[3]));
    __syncthreads();

    // --- segment sum of exp ---
    float sum = 0.f;
    for (int i = start + tid; i < end; i += HIDDEN) sum += __expf(s[i] - m);
    sum = waveSum(sum);
    if (lane == 0) red[wave] = sum;
    __syncthreads();
    sum = red[0] + red[1] + red[2] + red[3];
    const float inv = (sum > 0.f) ? 1.0f / sum : 0.f;

    // --- weighted pooling: thread tid owns output column tid ---
    float acc = 0.f;
    for (int base = start; base < end; base += HIDDEN) {
        __syncthreads();
        const int i = base + tid;
        wbuf[tid] = (i < end) ? __expf(s[i] - m) * inv : 0.f;
        __syncthreads();
        if (base + HIDDEN < end)                 // speculative prefetch of next chunk
            __builtin_prefetch(&h[(size_t)(base + HIDDEN) * HIDDEN + tid], 0, 0);
        const int cnt = (end - base < HIDDEN) ? (end - base) : HIDDEN;
#pragma unroll 4
        for (int j = 0; j < cnt; ++j)            // coalesced 512B read per iteration
            acc = fmaf(wbuf[j], h[(size_t)(base + j) * HIDDEN + tid], acc);
    }
    out[(size_t)g * HIDDEN + tid] = acc;         // empty graphs write 0
}

extern "C" void kernel_launch(void* const* d_in, const int* in_sizes, int n_in,
                              void* d_out, int out_size, void* d_ws, size_t ws_size,
                              hipStream_t stream) {
    const float*     h     = (const float*)d_in[0];
    const long long* batch = (const long long*)d_in[1];   // int64, sorted
    const float*     W1    = (const float*)d_in[2];
    const float*     b1    = (const float*)d_in[3];
    const float*     w2    = (const float*)d_in[4];
    const float*     b2    = (const float*)d_in[5];
    float* out = (float*)d_out;
    float* s   = (float*)d_ws;                            // N floats of scratch

    const int n          = in_sizes[0] / HIDDEN;          // number of nodes
    const int num_graphs = out_size / HIDDEN;
    const int tiles      = n >> 4;                        // full 16-row tiles

    if (tiles > 0) {
        int blocks_needed = (tiles + 7) / 8;              // 8 tiles per block pass
        int grid1 = blocks_needed < 3072 ? blocks_needed : 3072;
        score_kernel<<<grid1, 256, 0, stream>>>(h, W1, b1, w2, b2, s, n);
    }
    const int tail = n - (tiles << 4);
    if (tail > 0)
        score_tail_kernel<<<tail, HIDDEN, 0, stream>>>(h, W1, b1, w2, b2, s, tiles << 4);

    pool_kernel<<<num_graphs, HIDDEN, 0, stream>>>(h, batch, s, out, n);
}
